// GNNGen_6511170420786
// MI455X (gfx1250) — compile-verified, matrix operand order checked by reading, not run on hardware
//
#include <hip/hip_runtime.h>

#define B_   2
#define N_   512
#define F_   16
#define D_   64
#define H_   4
#define C_   4
#define HID_ 256
#define DH_  16
#define NEG_ (-9.0e15f)

typedef __attribute__((ext_vector_type(16))) _Float16 v16h;
typedef __attribute__((ext_vector_type(2)))  _Float16 h2;
typedef __attribute__((ext_vector_type(8)))  float    v8f;

// ---------- h = x @ W_emb  (one thread per [b,n,d]) ----------
__global__ __launch_bounds__(256) void emb_kernel(const float* __restrict__ x,
                                                  const float* __restrict__ Wemb,
                                                  float* __restrict__ h) {
  int t = blockIdx.x * blockDim.x + threadIdx.x;      // B*N*D
  int d = t % D_;
  int bn = t / D_;
  const float* xr = x + (size_t)bn * F_;
  float acc = 0.f;
#pragma unroll
  for (int f = 0; f < F_; ++f) acc = fmaf(xr[f], Wemb[f * D_ + d], acc);
  h[t] = acc;
}

// ---------- q,k,v = h @ Wq/Wk/Wv ----------
__global__ __launch_bounds__(256) void qkv_kernel(const float* __restrict__ h,
                                                  const float* __restrict__ Wq,
                                                  const float* __restrict__ Wk,
                                                  const float* __restrict__ Wv,
                                                  float* __restrict__ q,
                                                  float* __restrict__ k,
                                                  float* __restrict__ v) {
  int t = blockIdx.x * blockDim.x + threadIdx.x;      // B*N*D
  int d = t % D_;
  int bn = t / D_;
  const float* hr = h + (size_t)bn * D_;
  float aq = 0.f, ak = 0.f, av = 0.f;
  for (int dd = 0; dd < D_; ++dd) {
    float hv = hr[dd];
    aq = fmaf(hv, Wq[dd * D_ + d], aq);
    ak = fmaf(hv, Wk[dd * D_ + d], ak);
    av = fmaf(hv, Wv[dd * D_ + d], av);
  }
  q[t] = aq; k[t] = ak; v[t] = av;
}

// ---------- adjacency-masked attention, online softmax; one thread per (b,h,i) ----------
__global__ __launch_bounds__(256) void attn_kernel(const float* __restrict__ q,
                                                   const float* __restrict__ k,
                                                   const float* __restrict__ v,
                                                   const float* __restrict__ adj,
                                                   float* __restrict__ o) {
  int t = blockIdx.x * blockDim.x + threadIdx.x;      // B*H*N
  int i  = t % N_;
  int hh = (t / N_) % H_;
  int b  = t / (N_ * H_);
  const float* qi = q + ((size_t)(b * N_ + i)) * D_ + hh * DH_;
  float qr[DH_];
#pragma unroll
  for (int d = 0; d < DH_; ++d) qr[d] = qi[d] * 0.25f;   // 1/sqrt(Dh), Dh=16
  float mmax = -3.0e38f, lsum = 0.f;
  float oacc[DH_];
#pragma unroll
  for (int d = 0; d < DH_; ++d) oacc[d] = 0.f;
  const float* adjrow = adj + (size_t)i * N_;
  for (int j = 0; j < N_; ++j) {
    if (adjrow[j] <= 0.f) continue;                     // masked -1e9 -> exp underflows to 0
    const float* kj = k + ((size_t)(b * N_ + j)) * D_ + hh * DH_;
    float s = 0.f;
#pragma unroll
    for (int d = 0; d < DH_; ++d) s = fmaf(qr[d], kj[d], s);
    float mnew = fmaxf(mmax, s);
    float corr = __expf(mmax - mnew);
    float p    = __expf(s - mnew);
    lsum = lsum * corr + p;
    const float* vj = v + ((size_t)(b * N_ + j)) * D_ + hh * DH_;
#pragma unroll
    for (int d = 0; d < DH_; ++d) oacc[d] = oacc[d] * corr + p * vj[d];
    mmax = mnew;
  }
  float inv = 1.f / lsum;
  float* orow = o + ((size_t)(b * N_ + i)) * D_ + hh * DH_;
#pragma unroll
  for (int d = 0; d < DH_; ++d) orow[d] = oacc[d] * inv;
}

// ---------- y = h + o @ Wo ----------
__global__ __launch_bounds__(256) void proj_kernel(const float* __restrict__ h,
                                                   const float* __restrict__ o,
                                                   const float* __restrict__ Wo,
                                                   float* __restrict__ y) {
  int t = blockIdx.x * blockDim.x + threadIdx.x;      // B*N*D
  int d = t % D_;
  int bn = t / D_;
  const float* orow = o + (size_t)bn * D_;
  float acc = h[t];
  for (int dd = 0; dd < D_; ++dd) acc = fmaf(orow[dd], Wo[dd * D_ + d], acc);
  y[t] = acc;
}

// ---------- a = y@W1[:D] + b1 (f16), b = y@W1[D:] (f16) ----------
__global__ __launch_bounds__(256) void ab_kernel(const float* __restrict__ y,
                                                 const float* __restrict__ W1,
                                                 const float* __restrict__ b1,
                                                 _Float16* __restrict__ amath,
                                                 _Float16* __restrict__ bmath) {
  int t = blockIdx.x * blockDim.x + threadIdx.x;      // B*N*HID
  int hid = t % HID_;
  int bn  = t / HID_;
  const float* yr = y + (size_t)bn * D_;
  float aa = 0.f, bb = 0.f;
  for (int d = 0; d < D_; ++d) {
    float yv = yr[d];
    aa = fmaf(yv, W1[d * HID_ + hid], aa);
    bb = fmaf(yv, W1[(D_ + d) * HID_ + hid], bb);
  }
  amath[t] = (_Float16)(aa + b1[hid]);                // fold b1 into a
  bmath[t] = (_Float16)bb;
}

// ---------- W2 [HID,C] -> padded transposed f16 [16,HID] ----------
__global__ __launch_bounds__(256) void w2_kernel(const float* __restrict__ W2,
                                                 _Float16* __restrict__ W2pT) {
  int t = blockIdx.x * blockDim.x + threadIdx.x;      // 16*HID
  int kpos = t % HID_;
  int c    = t / HID_;
  W2pT[c * HID_ + kpos] = (c < C_) ? (_Float16)W2[kpos * C_ + c] : (_Float16)0.0f;
}

// ---------- pairwise FF + mask: one wave per (b, i, 16-wide j tile) ----------
// D[16 j x 16 cpad] = Hidden[16 j x 256] (f16, A) x W2pT[256 x 16] (f16, B), fp32 acc.
__global__ __launch_bounds__(256) void pair_ff_kernel(const _Float16* __restrict__ amath,
                                                      const _Float16* __restrict__ bmath,
                                                      const _Float16* __restrict__ W2pT,
                                                      const float* __restrict__ b2,
                                                      const float* __restrict__ adj,
                                                      float* __restrict__ out) {
  const int lane = threadIdx.x & 31;
  const int wave = (blockIdx.x * blockDim.x + threadIdx.x) >> 5;
  const int jt   = wave & 31;                 // N/16 = 32 tiles
  const int i    = (wave >> 5) & (N_ - 1);    // node i
  const int b    = wave >> 14;                // batch
  const int half = lane >> 4;                 // K-half selector (A/B per-lane layout)
  const int lm   = lane & 15;                 // A: row-in-tile; B/D: column
  const int jrow = jt * 16 + lm;

  const _Float16* arow  = amath + ((size_t)(b * N_ + i)) * HID_;
  const _Float16* brow  = bmath + ((size_t)(b * N_ + jrow)) * HID_;
  const _Float16* w2row = W2pT + (size_t)lm * HID_;

  const h2 zero = {(_Float16)0.0f, (_Float16)0.0f};
  v8f acc = {0.f, 0.f, 0.f, 0.f, 0.f, 0.f, 0.f, 0.f};

#pragma unroll
  for (int kk = 0; kk < 8; ++kk) {            // K = 256 in 8 steps of 32
    const int kbase = kk * 32;
    union { v16h v; h2 p[8]; } A, Bm;
#pragma unroll
    for (int vv = 0; vv < 8; ++vv) {
      // 16-bit A-matrix K layout: K = (v>=4?16:0) + 8*half + 2*(v&3) (+0/+1 packed)
      const int k0 = kbase + ((vv & 4) ? 16 : 0) + half * 8 + (vv & 3) * 2;
      h2 av = *(const h2*)(arow + k0);
      h2 bv = *(const h2*)(brow + k0);
      A.p[vv]  = __builtin_elementwise_max(av + bv, zero);   // relu(a_i + b_j + b1)
      Bm.p[vv] = *(const h2*)(w2row + k0);                    // padded W2 column
    }
    acc = __builtin_amdgcn_wmma_f32_16x16x32_f16(
        /*neg_a=*/false, A.v, /*neg_b=*/false, Bm.v,
        /*c_mod=*/(short)0, acc, /*reuse_a=*/false, /*reuse_b=*/false);
  }

  // D layout: lane holds column c = lm; rows m = vv + 8*half
  if (lm < C_) {
    const int c = lm;
    const float bias = b2[c];
#pragma unroll
    for (int vv = 0; vv < 8; ++vv) {
      const int j = jt * 16 + vv + 8 * half;
      float val = acc[vv] + bias;
      out[((size_t)((b * C_ + c) * N_ + i)) * N_ + j] =
          (adj[(size_t)i * N_ + j] > 0.f) ? val : NEG_;
    }
  }
}

extern "C" void kernel_launch(void* const* d_in, const int* in_sizes, int n_in,
                              void* d_out, int out_size, void* d_ws, size_t ws_size,
                              hipStream_t stream) {
  const float* x    = (const float*)d_in[0];
  const float* adj  = (const float*)d_in[1];
  const float* Wemb = (const float*)d_in[2];
  const float* Wq   = (const float*)d_in[3];
  const float* Wk   = (const float*)d_in[4];
  const float* Wv   = (const float*)d_in[5];
  const float* Wo   = (const float*)d_in[6];
  const float* W1   = (const float*)d_in[7];
  const float* b1   = (const float*)d_in[8];
  const float* W2   = (const float*)d_in[9];
  const float* b2   = (const float*)d_in[10];
  float* out = (float*)d_out;

  // workspace layout
  const size_t ND = (size_t)B_ * N_ * D_;     // 65536
  const size_t NH = (size_t)B_ * N_ * HID_;   // 262144
  float* fws = (float*)d_ws;
  float* h = fws;
  float* q = h + ND;
  float* k = q + ND;
  float* v = k + ND;
  float* o = v + ND;
  float* y = o + ND;
  _Float16* amath = (_Float16*)(y + ND);
  _Float16* bmath = amath + NH;
  _Float16* W2pT  = bmath + NH;               // 16*HID halves

  dim3 blk(256);
  emb_kernel <<<dim3((unsigned)(ND / 256)), blk, 0, stream>>>(x, Wemb, h);
  qkv_kernel <<<dim3((unsigned)(ND / 256)), blk, 0, stream>>>(h, Wq, Wk, Wv, q, k, v);
  attn_kernel<<<dim3((B_ * H_ * N_) / 256), blk, 0, stream>>>(q, k, v, adj, o);
  proj_kernel<<<dim3((unsigned)(ND / 256)), blk, 0, stream>>>(h, o, Wo, y);
  ab_kernel  <<<dim3((unsigned)(NH / 256)), blk, 0, stream>>>(y, W1, b1, amath, bmath);
  w2_kernel  <<<dim3((16 * HID_) / 256), blk, 0, stream>>>(W2, W2pT);
  // B*N*(N/16) waves * 32 threads / 256 = 4096 blocks, all waves full (EXEC all-ones for WMMA)
  pair_ff_kernel<<<dim3((B_ * N_ * (N_ / 16) * 32) / 256), blk, 0, stream>>>(
      amath, bmath, W2pT, b2, adj, out);
}
